// DiffusionGraphConv_84576495993174
// MI455X (gfx1250) — matrix operand order repaired and checked by reading, not compiled, not run on hardware
//
#include <hip/hip_runtime.h>
#include <hip/hip_bf16.h>

// ---------------- problem constants (match reference) ----------------
#define NN    8192      // nodes
#define BB    32        // batch
#define FF    128       // input_dim + hid_dim
#define BF    4096      // BB * FF  (folded batch*feature dim for SpMM)
#define EE    262144    // edges per support
#define MM    5         // num_matrices = 2 supports * K(2) + 1
#define KDIM  640       // FF * MM  (GEMM reduction dim)
#define ODIM  64        // output dim
#define ROWS  262144    // BB * NN  (GEMM M dim)

typedef __attribute__((ext_vector_type(16))) __bf16          v16bf;
typedef __attribute__((ext_vector_type(8)))  float           v8f;
typedef __attribute__((ext_vector_type(8)))  unsigned short  u16x8;
typedef __attribute__((ext_vector_type(4)))  unsigned int    u32x4;
typedef __attribute__((ext_vector_type(4)))  int             i32x4;
typedef __attribute__((ext_vector_type(8)))  int             i32x8;

__device__ __forceinline__ unsigned short f2bfu(float f) {
    unsigned u = __float_as_uint(f);
    unsigned r = u + 0x7FFFu + ((u >> 16) & 1u);   // round-to-nearest-even
    return (unsigned short)(r >> 16);
}

// ---------------- phase 1: X0 = concat(inputs, state) laid out [N, B*F] ----
__global__ void concat_x0(const float* __restrict__ inp,
                          const float* __restrict__ st,
                          float* __restrict__ x0) {
    long i = (long)blockIdx.x * blockDim.x + threadIdx.x;   // i < NN*BF
    int n = (int)(i >> 12);
    int r = (int)(i & 4095);
    int b = r >> 7;
    int f = r & 127;
    long src = ((long)b * NN + n) * 64;
    x0[i] = (f < 64) ? inp[src + f] : st[src + (f - 64)];
}

// ---------------- helpers: zero / x2 := -x0 (float4 streams) ---------------
__global__ void zero_f4(float4* __restrict__ p) {
    long i = (long)blockIdx.x * blockDim.x + threadIdx.x;
    float4 z; z.x = 0.f; z.y = 0.f; z.z = 0.f; z.w = 0.f;
    p[i] = z;
}

__global__ void negcopy_f4(const float4* __restrict__ s, float4* __restrict__ d) {
    long i = (long)blockIdx.x * blockDim.x + threadIdx.x;
    float4 v = s[i];
    v.x = -v.x; v.y = -v.y; v.z = -v.z; v.w = -v.w;
    d[i] = v;
}

// ---------------- phase 2: SpMM scatter-add, one block per edge ------------
// xout[row,:] += scale * val(e) * xin[col,:]   (4096 cols, coalesced)
__global__ void spmm_atomic(const int*   __restrict__ rows,
                            const int*   __restrict__ cols,
                            const float* __restrict__ vals,
                            const float* __restrict__ xin,
                            float*       __restrict__ xout,
                            float scale) {
    int e   = blockIdx.x;
    int row = rows[e];
    int col = cols[e];
    float v = vals[e] * scale;
    const float* src = xin  + (long)col * BF;
    float*       dst = xout + (long)row * BF;
    for (int c = threadIdx.x; c < BF; c += blockDim.x)
        atomicAdd(dst + c, v * src[c]);          // global_atomic_add_f32 (no return)
}

// ---------------- phase 3: pack A to interleaved bf16 [ROWS, KDIM] ---------
// A[b*N+n, f*5+m] = xs[m][n*BF + b*FF + f]   (reference permute(0,2,3,1))
__global__ void pack_a(const float* __restrict__ x0,  const float* __restrict__ x1a,
                       const float* __restrict__ x2a, const float* __restrict__ x1b,
                       const float* __restrict__ x2b, unsigned short* __restrict__ ap) {
    long i   = (long)blockIdx.x * blockDim.x + threadIdx.x;  // i < ROWS*FF
    long row = i >> 7;
    int  f   = (int)(i & 127);
    int  b   = (int)(row >> 13);      // row = b*NN + n
    int  n   = (int)(row & 8191);
    long off = (long)n * BF + (long)b * FF + f;
    unsigned short* o = ap + row * KDIM + (long)f * MM;
    o[0] = f2bfu(x0[off]);
    o[1] = f2bfu(x1a[off]);
    o[2] = f2bfu(x2a[off]);
    o[3] = f2bfu(x1b[off]);
    o[4] = f2bfu(x2b[off]);
}

// weight [KDIM, ODIM] f32 -> Wt [ODIM, KDIM] bf16 (K-contiguous per column)
__global__ void pack_w(const float* __restrict__ w, unsigned short* __restrict__ wt) {
    int i = blockIdx.x * blockDim.x + threadIdx.x;   // i < KDIM*ODIM
    int k = i >> 6;
    int c = i & 63;
    wt[(long)c * KDIM + k] = f2bfu(w[(long)k * ODIM + c]);
}

// ---------------- phase 4: GEMM via v_wmma_f32_16x16x32_bf16 ---------------
// 8 waves/block; each wave computes one 16-row strip x all 64 output columns
// (4 accumulators) so every A fragment is reused 4x. The full bf16 weight
// (640x64 = 80 KB) is staged once into LDS — via the Tensor Data Mover when
// the builtin is available (one TENSOR_LOAD_TO_LDS, TENSORcnt-tracked),
// otherwise via cooperative b128 copies.
__global__ void __launch_bounds__(256)
gemm_wmma(const unsigned short* __restrict__ A,
          const unsigned short* __restrict__ Wt,
          const float* __restrict__ bias,
          float* __restrict__ out) {
    extern __shared__ unsigned short lds[];          // KDIM*ODIM ushorts = 80 KB

#if __has_builtin(__builtin_amdgcn_tensor_load_to_lds)
    // ---- TDM stage: one descriptor copies the whole 81920-byte weight ----
    if ((threadIdx.x >> 5) == 0) {                   // one TDM op per block
        unsigned long long ga = (unsigned long long)(uintptr_t)Wt;
        unsigned lds_addr     = (unsigned)(uintptr_t)lds;   // LDS byte offset
        const unsigned NELEM  = KDIM * ODIM;                // 40960 x 2-byte
        // D# group 0: count=1 | lds_addr | global_addr[56:0] | type=2
        u32x4 g0;
        g0[0] = 1u;                                  // count=1, is_restore=0
        g0[1] = lds_addr;                            // bits 63:32 lds_addr
        g0[2] = (unsigned)(ga & 0xFFFFFFFFu);        // global_addr low
        g0[3] = (unsigned)((ga >> 32) & 0x01FFFFFFu) | (2u << 30);  // type=2
        // D# group 1: data_size=1 (2B); 1-D tensor_dim0=tile_dim0=NELEM
        i32x8 g1;
        g1[0] = (int)(1u << 16);                     // data_size=1 (2 bytes)
        g1[1] = (int)((NELEM & 0xFFFFu) << 16);      // tensor_dim0[15:0]
        g1[2] = (int)(((NELEM >> 16) & 0xFFFFu) |    // tensor_dim0[31:16]
                      (1u << 16));                   // tensor_dim1 = 1
        g1[3] = (int)((NELEM & 0xFFFFu) << 16);      // tile_dim0 = NELEM
        g1[4] = 0;                                   // tile_dim1/2 unused
        g1[5] = (int)NELEM;                          // tensor_dim0_stride lo
        g1[6] = 0;
        g1[7] = 0;
        i32x4 gz = {0, 0, 0, 0};                     // groups 2/3: <=2D tensor
#if __has_include(<hip/amd_detail/amd_gfx1250_TDM.h>)
        i32x8 gz8 = {0, 0, 0, 0, 0, 0, 0, 0};
        __builtin_amdgcn_tensor_load_to_lds(g0, g1, gz, gz, gz8, 0);
#else
        __builtin_amdgcn_tensor_load_to_lds(g0, g1, gz, gz, 0);
#endif
        __builtin_amdgcn_s_wait_tensorcnt(0);        // TENSORcnt == 0
    }
#else
    // ---- fallback: cooperative b128 copy of the weight into LDS ----------
    {
        const u16x8* src = (const u16x8*)Wt;
        u16x8*       dst = (u16x8*)lds;
        const int nvec = (KDIM * ODIM) / 8;          // 5120
        for (int i = threadIdx.x; i < nvec; i += 256)
            dst[i] = src[i];
    }
#endif
    __syncthreads();

    int wave = threadIdx.x >> 5;
    int lane = threadIdx.x & 31;
    long rowTile = (long)blockIdx.x * 8 + wave;
    long rowBase = rowTile * 16;
    int  half = lane >> 4;          // 0 or 1
    int  r    = lane & 15;

    const unsigned short* arow = A + (rowBase + r) * KDIM + half * 8;

    v8f acc0 = {}, acc1 = {}, acc2 = {}, acc3 = {};
    #pragma unroll 2
    for (int k = 0; k < KDIM; k += 32) {
        union { u16x8 u[2]; v16bf v; } a;
        const u16x8* ap = (const u16x8*)(arow + k);
        a.u[0] = ap[0];     // K = kb+0..7   (or kb+8..15 for upper half-wave)
        a.u[1] = ap[2];     // K = kb+16..23 (or kb+24..31)

        const unsigned short* bbase = lds + r * KDIM + k + half * 8;
        #pragma unroll
        for (int ct = 0; ct < 4; ++ct) {
            union { u16x8 u[2]; v16bf v; } b;
            const u16x8* bp = (const u16x8*)(bbase + ct * (16 * KDIM));
            b.u[0] = bp[0]; // K = kb+0..7 / kb+8..15
            b.u[1] = bp[2];
            v8f c = (ct == 0) ? acc0 : (ct == 1) ? acc1 : (ct == 2) ? acc2 : acc3;
            c = __builtin_amdgcn_wmma_f32_16x16x32_bf16(
                    false, a.v, false, b.v, (short)0, c, false, false);
            if      (ct == 0) acc0 = c;
            else if (ct == 1) acc1 = c;
            else if (ct == 2) acc2 = c;
            else              acc3 = c;
        }
    }

    // ---- epilogue: bias + store -------------------------------------------
    #pragma unroll
    for (int ct = 0; ct < 4; ++ct) {
        v8f acc = (ct == 0) ? acc0 : (ct == 1) ? acc1 : (ct == 2) ? acc2 : acc3;
        int   col = ct * 16 + r;
        float bv  = bias[col];
        float* obase = out + (rowBase + 8 * half) * ODIM + col;
        #pragma unroll
        for (int i = 0; i < 8; ++i)
            obase[(long)i * ODIM] = acc[i] + bv;
    }
}

// ---------------- host launcher --------------------------------------------
extern "C" void kernel_launch(void* const* d_in, const int* in_sizes, int n_in,
                              void* d_out, int out_size, void* d_ws, size_t ws_size,
                              hipStream_t stream) {
    const float* inputs = (const float*)d_in[0];
    const float* state  = (const float*)d_in[1];
    const int*   rows   = (const int*)  d_in[2];   // [2, E]
    const int*   cols   = (const int*)  d_in[3];   // [2, E]
    const float* vals   = (const float*)d_in[4];   // [2, E]
    const float* weight = (const float*)d_in[5];   // [640, 64]
    const float* biases = (const float*)d_in[6];   // [64]
    float* out = (float*)d_out;

    char* ws = (char*)d_ws;
    const size_t S = (size_t)NN * BF * sizeof(float);     // 128 MB per matrix
    float* X0  = (float*)(ws + 0 * S);
    float* X1a = (float*)(ws + 1 * S);
    float* X2a = (float*)(ws + 2 * S);
    float* X1b = (float*)(ws + 3 * S);
    float* X2b = (float*)(ws + 4 * S);
    unsigned short* Ap = (unsigned short*)(ws + 5 * S);                       // 335 MB
    unsigned short* Wt = (unsigned short*)(ws + 5 * S +
                                           (size_t)ROWS * KDIM * sizeof(unsigned short));

    const int T = 256;
    const long elemsX  = (long)NN * BF;        // 33.5M
    const long f4X     = elemsX / 4;

    // 1) X0 = concat(inputs, state)
    concat_x0<<<(unsigned)(elemsX / T), T, 0, stream>>>(inputs, state, X0);

    // 2) support 0: x1 = A@x0 ; x2 = 2*A@x1 - x0
    zero_f4<<<(unsigned)(f4X / T), T, 0, stream>>>((float4*)X1a);
    spmm_atomic<<<EE, T, 0, stream>>>(rows, cols, vals, X0, X1a, 1.0f);
    negcopy_f4<<<(unsigned)(f4X / T), T, 0, stream>>>((const float4*)X0, (float4*)X2a);
    spmm_atomic<<<EE, T, 0, stream>>>(rows, cols, vals, X1a, X2a, 2.0f);

    //    support 1
    zero_f4<<<(unsigned)(f4X / T), T, 0, stream>>>((float4*)X1b);
    spmm_atomic<<<EE, T, 0, stream>>>(rows + EE, cols + EE, vals + EE, X0, X1b, 1.0f);
    negcopy_f4<<<(unsigned)(f4X / T), T, 0, stream>>>((const float4*)X0, (float4*)X2b);
    spmm_atomic<<<EE, T, 0, stream>>>(rows + EE, cols + EE, vals + EE, X1b, X2b, 2.0f);

    // 3) pack operands to bf16
    pack_a<<<(unsigned)(((long)ROWS * FF) / T), T, 0, stream>>>(X0, X1a, X2a, X1b, X2b, Ap);
    pack_w<<<(KDIM * ODIM) / T, T, 0, stream>>>(weight, Wt);

    // 4) WMMA GEMM: [262144, 640] x [640, 64] + bias; 80 KB LDS for the weight
    gemm_wmma<<<ROWS / 16 / 8, T, KDIM * ODIM * sizeof(unsigned short), stream>>>(
        Ap, Wt, biases, out);
}
